// TripleAttentionSmarter_21809843929839
// MI455X (gfx1250) — matrix operand order repaired and checked by reading, not compiled, not run on hardware
//
#include <hip/hip_runtime.h>

// ---------------------------------------------------------------------------
// TripleAttentionSmarter forward for MI455X (gfx1250).
// B=4, N=2048, C=512, H=1, D=512.  ~253 GFLOP, dominated by two N^3 GEMMs.
//
// All GEMMs run on v_wmma_f32_16x16x32_f16 (f16 in, f32 accumulate).
// f16 intermediates keep X/Y/Z/M1T/M2 (160MB) ~resident in the 192MB L2.
// Tile staging uses the CDNA5 Tensor Data Mover (tensor_load_to_lds, TENSORcnt)
// with double-buffered LDS so DMA for K-step s+1 overlaps WMMA of step s;
// transposed operands use register-staged transpose scatter instead.
//
// Pipeline:
//  K0  cast x, w_weight, o_weight to f16
//  K1  P = x @ Ww^T                   [8192,2560] f16  (a|b|c|v1|v2 blocks)
//  K2  X=exp(s*a@b^T), Y=exp(s*b@c^T), Z=exp(s*c@a^T)  f16 [B,2048,2048]
//  K3a M1T = (Y@Z) .* X^T             f16
//  K3b M2  = (X@Y) .* Z^T             f16
//  K4d den[i] = colsum_j M1T[j,i]     f32
//  K4a U[:, :512] = (M1T^T @ v1)/den  f16
//  K4b U[:, 512:] = (M2   @ v2)/den   f16
//  K5  out = U @ Ow^T + bias          f32 -> d_out
// Workspace: ~227.5 MiB.
// ---------------------------------------------------------------------------

typedef __attribute__((ext_vector_type(16))) _Float16 v16h;
typedef __attribute__((ext_vector_type(8)))  _Float16 v8h;
typedef __attribute__((ext_vector_type(8)))  float    v8f;
typedef __attribute__((ext_vector_type(4)))  unsigned int u32x4;
typedef __attribute__((ext_vector_type(8)))  int i32x8;
typedef __attribute__((ext_vector_type(4)))  int i32x4;

#if defined(__HIP_DEVICE_COMPILE__) && __has_builtin(__builtin_amdgcn_tensor_load_to_lds) && __has_builtin(__builtin_amdgcn_s_wait_tensorcnt)
#define USE_TDM 1
#else
#define USE_TDM 0
#endif

#define LDS_S64 40  // halves stride: 64B row + 16B pad = 80B (16B aligned)

struct Regs { v8h r0, r1; };

// ---- register-staged copies (straight-line, 2 chunks/thread, 128 threads) ----
__device__ __forceinline__ Regs gload_64x32(const _Float16* src, int ld) {
  int t = (int)threadIdx.x, c0 = t, c1 = t + 128;
  Regs R;
  R.r0 = *(const v8h*)(src + (long long)(c0 >> 2) * ld + (c0 & 3) * 8);
  R.r1 = *(const v8h*)(src + (long long)(c1 >> 2) * ld + (c1 & 3) * 8);
  return R;
}
__device__ __forceinline__ void store_64x32(_Float16* dst, Regs R) {
  int t = (int)threadIdx.x, c0 = t, c1 = t + 128;
  *(v8h*)(dst + (c0 >> 2) * LDS_S64 + (c0 & 3) * 8) = R.r0;
  *(v8h*)(dst + (c1 >> 2) * LDS_S64 + (c1 & 3) * 8) = R.r1;
}
// transposed operand: source tile is 32 rows(k) x 64 cols(m|n); LDS stays
// fragment-major [m][k] so fragment reads remain ds_load_b128.
__device__ __forceinline__ Regs gload_32x64(const _Float16* src, int ld) {
  int t = (int)threadIdx.x, c0 = t, c1 = t + 128;
  Regs R;
  R.r0 = *(const v8h*)(src + (long long)(c0 >> 3) * ld + (c0 & 7) * 8);
  R.r1 = *(const v8h*)(src + (long long)(c1 >> 3) * ld + (c1 & 7) * 8);
  return R;
}
__device__ __forceinline__ void store_32x64_tr(_Float16* dst, Regs R) {
  int t = (int)threadIdx.x, c0 = t, c1 = t + 128;
  int k0 = c0 >> 3, m0 = (c0 & 7) * 8;
  int k1 = c1 >> 3, m1 = (c1 & 7) * 8;
  #pragma unroll
  for (int j = 0; j < 8; ++j) {
    dst[(m0 + j) * LDS_S64 + k0] = R.r0[j];
    dst[(m1 + j) * LDS_S64 + k1] = R.r1[j];
  }
}

#if USE_TDM
// TDM: DMA a 64-row x 32-half tile (row stride `ld` halves) into LDS with
// hardware padding: 4 dwords pad after every 16 dwords -> 80B LDS rows.
// D# bitfields per CDNA5 ISA ch.8 (group0: flags/lds/global addr, group1:
// sizes/strides/pad). 2D tile -> extension groups zero.
__device__ __forceinline__ void tdm_tile64x32(const _Float16* gsrc, _Float16* lds_dst, int ld) {
  unsigned long long ga = (unsigned long long)(size_t)gsrc;
  unsigned lds = (unsigned)(size_t)(void*)lds_dst;  // flat addr[31:0] == LDS offset
  u32x4 g0;
  g0[0] = 1u;                                         // count=1, user descriptor
  g0[1] = lds;                                        // lds_addr
  g0[2] = (unsigned)ga;                               // global_addr[31:0]
  g0[3] = (unsigned)((ga >> 32) & 0x01FFFFFFull) | (2u << 30);  // ga[56:32] | type=2
  const unsigned td0 = 0x40000000u, td1 = 0x40000000u;  // huge tensor bounds (no OOB clip)
  i32x8 g1;
  g1[0] = (int)((1u << 16)          // data_size = 2 bytes
              | (1u << 20)          // pad_enable
              | (3u << 22)          // pad_interval: 16 dwords (64B)
              | (3u << 25));        // pad_amount:   4 dwords (16B)
  g1[1] = (int)((td0 & 0xFFFFu) << 16);                         // tensor_dim0 lo
  g1[2] = (int)((td0 >> 16) | ((td1 & 0xFFFFu) << 16));         // td0 hi | td1 lo
  g1[3] = (int)(((td1 >> 16) & 0xFFFFu) | (32u << 16));         // td1 hi | tile_dim0=32
  g1[4] = 64;                                                   // tile_dim1=64
  g1[5] = (int)(unsigned)ld;                                    // tensor_dim0_stride lo
  g1[6] = 0;                                                    // stride hi | dim1_stride lo
  g1[7] = 0;
  i32x4 z4; z4[0] = z4[1] = z4[2] = z4[3] = 0;
  i32x8 z8; z8[0] = z8[1] = z8[2] = z8[3] = z8[4] = z8[5] = z8[6] = z8[7] = 0;
  __builtin_amdgcn_tensor_load_to_lds(g0, g1, z4, z4, z8, 0);
}
#endif

// EPI: 0 = f32 store + bias (aux=f32 bias)
//      1 = f16 store
//      2 = f16 store of exp(acc*scale)
//      3 = f16 store of acc * aux^T   (aux = f16 matrix, element [gn][gm])
//      4 = f16 store of acc / aux[gm] (aux = f32 per-row denominator)
template <int EPI, bool AT, bool BT>
__global__ __launch_bounds__(128) void gemm_wmma(
    const _Float16* __restrict__ A, long long sA, int lda,
    const _Float16* __restrict__ Bm, long long sB, int ldb,
    void* __restrict__ Cv, long long sC, int ldc,
    const void* __restrict__ auxv, long long sAux, int ldaux,
    int K, float scale) {
  __shared__ _Float16 As[2][64 * LDS_S64];
  __shared__ _Float16 Bs[2][64 * LDS_S64];

  const int b  = (int)blockIdx.z;
  const int m0 = (int)blockIdx.y * 64;
  const int n0 = (int)blockIdx.x * 64;
  A  += (long long)b * sA;
  Bm += (long long)b * sB;

  const int lane = (int)threadIdx.x & 31;
  const int wave = (int)threadIdx.x >> 5;
  const bool w0  = (wave == 0);
  const int nsteps = K / 32;

  v8f acc[4] = {};
  Regs ra, rb;

  // ---- prologue: stage K-tile 0 into buffer 0 ----
#if USE_TDM
  if (!AT) { if (w0) tdm_tile64x32(A + (long long)m0 * lda, &As[0][0], lda); }
  else     { ra = gload_32x64(A + m0, lda); store_32x64_tr(&As[0][0], ra); }
  if (BT)  { if (w0) tdm_tile64x32(Bm + (long long)n0 * ldb, &Bs[0][0], ldb); }
  else     { rb = gload_32x64(Bm + n0, ldb); store_32x64_tr(&Bs[0][0], rb); }
  if (w0 && (!AT || BT)) __builtin_amdgcn_s_wait_tensorcnt(0);
#else
  if (!AT) { ra = gload_64x32(A + (long long)m0 * lda, lda); store_64x32(&As[0][0], ra); }
  else     { ra = gload_32x64(A + m0, lda); store_32x64_tr(&As[0][0], ra); }
  if (BT)  { rb = gload_64x32(Bm + (long long)n0 * ldb, ldb); store_64x32(&Bs[0][0], rb); }
  else     { rb = gload_32x64(Bm + n0, ldb); store_32x64_tr(&Bs[0][0], rb); }
#endif
  __syncthreads();

  int buf = 0;
  for (int s = 0; s < nsteps; ++s) {
    const bool more = (s + 1 < nsteps);
    const int kn = (s + 1) * 32;
    if (more) {
      // issue next stage into buf^1 (safe: last read barrier-separated)
#if USE_TDM
      if (!AT) { if (w0) tdm_tile64x32(A + (long long)m0 * lda + kn, &As[buf ^ 1][0], lda); }
      else     ra = gload_32x64(A + (long long)kn * lda + m0, lda);
      if (BT)  { if (w0) tdm_tile64x32(Bm + (long long)n0 * ldb + kn, &Bs[buf ^ 1][0], ldb); }
      else     rb = gload_32x64(Bm + (long long)kn * ldb + n0, ldb);
#else
      ra = AT ? gload_32x64(A + (long long)kn * lda + m0, lda)
              : gload_64x32(A + (long long)m0 * lda + kn, lda);
      rb = BT ? gload_64x32(Bm + (long long)n0 * ldb + kn, ldb)
              : gload_32x64(Bm + (long long)kn * ldb + n0, ldb);
#endif
    }

    // ---- compute from buffer `buf` (DMA/loads for s+1 in flight) ----
    const _Float16* as = &As[buf][0];
    const _Float16* bs = &Bs[buf][0];
    // Load ALL fragments first so a single s_wait_dscnt gates the first WMMA
    // and the remaining three issue back-to-back.
    // A fragment 16x32 f16 (lanes 0-15: M=lane, K=0..15; 16-31: K=16..31)
    v16h af;
    {
      int m  = lane & 15;
      int kb = (lane & 16) ? 16 : 0;
      int mm = wave * 16 + m;
      #pragma unroll
      for (int g = 0; g < 8; ++g) {
        int k = kb + 2 * g;
        af[2 * g]     = as[mm * LDS_S64 + k];
        af[2 * g + 1] = as[mm * LDS_S64 + k + 1];
      }
    }
    // B fragments 32x16 (lanes 0-15: N=lane, K=0..15; 16-31: K=16..31),
    // contiguous in LDS for every operand kind -> ds_load_b128 pairs
    v16h bf[4];
    {
      int kb2 = (lane & 16) ? 16 : 0;
      #pragma unroll
      for (int t = 0; t < 4; ++t) {
        int nn = t * 16 + (lane & 15);
        #pragma unroll
        for (int h = 0; h < 16; ++h) bf[t][h] = bs[nn * LDS_S64 + kb2 + h];
      }
    }
    #pragma unroll
    for (int t = 0; t < 4; ++t)
      acc[t] = __builtin_amdgcn_wmma_f32_16x16x32_f16(
          false, af, false, bf[t], (short)0, acc[t], false, false);

    if (more) {
      // drain next-stage staging, then publish
#if USE_TDM
      if (AT)  store_32x64_tr(&As[buf ^ 1][0], ra);
      if (!BT) store_32x64_tr(&Bs[buf ^ 1][0], rb);
      if (w0 && (!AT || BT)) __builtin_amdgcn_s_wait_tensorcnt(0);
#else
      if (AT) store_32x64_tr(&As[buf ^ 1][0], ra); else store_64x32(&As[buf ^ 1][0], ra);
      if (BT) store_64x32(&Bs[buf ^ 1][0], rb); else store_32x64_tr(&Bs[buf ^ 1][0], rb);
#endif
      __syncthreads();
      buf ^= 1;
    }
  }

  // C/D layout: VGPR r -> M = r (lanes 0-15) / r+8 (lanes 16-31), N = lane&15
  const int n_l   = lane & 15;
  const int m_off = (lane & 16) ? 8 : 0;
  #pragma unroll
  for (int t = 0; t < 4; ++t) {
    #pragma unroll
    for (int r = 0; r < 8; ++r) {
      int gm = m0 + wave * 16 + m_off + r;
      int gn = n0 + t * 16 + n_l;
      float v = acc[t][r];
      long long cidx = (long long)b * sC + (long long)gm * ldc + gn;
      if (EPI == 0) {
        const float* bias = (const float*)auxv;
        ((float*)Cv)[cidx] = v + bias[gn];
      } else if (EPI == 1) {
        ((_Float16*)Cv)[cidx] = (_Float16)v;
      } else if (EPI == 2) {
        ((_Float16*)Cv)[cidx] = (_Float16)__expf(v * scale);
      } else if (EPI == 3) {
        const _Float16* E = (const _Float16*)auxv + (long long)b * sAux;
        ((_Float16*)Cv)[cidx] = (_Float16)(v * (float)E[(long long)gn * ldaux + gm]);
      } else if (EPI == 4) {
        const float* den = (const float*)auxv + (long long)b * sAux;
        ((_Float16*)Cv)[cidx] = (_Float16)(v / den[gm]);
      }
    }
  }
}

__global__ __launch_bounds__(256) void cast_f32_f16(const float* __restrict__ src,
                                                    _Float16* __restrict__ dst, int n) {
  int i = (int)blockIdx.x * 256 + (int)threadIdx.x;
  if (i < n) dst[i] = (_Float16)src[i];
}

// den[b][i] = sum_j M1T[b][j][i]  (column sums, coalesced across i)
__global__ __launch_bounds__(256) void den_kernel(const _Float16* __restrict__ M1T,
                                                  float* __restrict__ den) {
  int b = (int)blockIdx.y;
  int i = (int)blockIdx.x * 256 + (int)threadIdx.x;
  const _Float16* p = M1T + (long long)b * 2048 * 2048 + i;
  float s = 0.f;
  for (int j = 0; j < 2048; ++j) s += (float)p[(long long)j * 2048];
  den[b * 2048 + i] = s;
}

extern "C" void kernel_launch(void* const* d_in, const int* in_sizes, int n_in,
                              void* d_out, int out_size, void* d_ws, size_t ws_size,
                              hipStream_t stream) {
  (void)in_sizes; (void)n_in; (void)out_size; (void)ws_size;

  const float* x  = (const float*)d_in[0];  // [4,2048,512]
  const float* ww = (const float*)d_in[1];  // [2560,512]
  const float* ow = (const float*)d_in[2];  // [512,1024]
  const float* ob = (const float*)d_in[3];  // [512]
  float* out = (float*)d_out;               // [4,2048,512]

  char* base = (char*)d_ws;
  size_t off = 0;
  auto carve = [&](size_t bytes) -> char* {
    char* p = base + off;
    off += (bytes + 255) & ~(size_t)255;
    return p;
  };
  _Float16* Xf16  = (_Float16*)carve((size_t)8192 * 512 * 2);
  _Float16* Wf16  = (_Float16*)carve((size_t)2560 * 512 * 2);
  _Float16* OWf16 = (_Float16*)carve((size_t)512 * 1024 * 2);
  _Float16* P     = (_Float16*)carve((size_t)8192 * 2560 * 2);
  _Float16* Xm    = (_Float16*)carve((size_t)4 * 2048 * 2048 * 2);
  _Float16* Ym    = (_Float16*)carve((size_t)4 * 2048 * 2048 * 2);
  _Float16* Zm    = (_Float16*)carve((size_t)4 * 2048 * 2048 * 2);
  _Float16* M1T   = (_Float16*)carve((size_t)4 * 2048 * 2048 * 2);
  _Float16* M2    = (_Float16*)carve((size_t)4 * 2048 * 2048 * 2);
  float*    den   = (float*)   carve((size_t)4 * 2048 * 4);
  _Float16* U     = (_Float16*)carve((size_t)8192 * 1024 * 2);

  const long long SP = 2048LL * 2560;
  const long long SN = 2048LL * 2048;
  const float scale = 0.044194173824159216f;  // 512^-0.5
  const dim3 blk(128);

  cast_f32_f16<<<dim3((8192 * 512) / 256), dim3(256), 0, stream>>>(x,  Xf16,  8192 * 512);
  cast_f32_f16<<<dim3((2560 * 512) / 256), dim3(256), 0, stream>>>(ww, Wf16,  2560 * 512);
  cast_f32_f16<<<dim3((512 * 1024) / 256), dim3(256), 0, stream>>>(ow, OWf16, 512 * 1024);

  // K1: P = x @ Ww^T
  gemm_wmma<1, false, true><<<dim3(40, 128, 1), blk, 0, stream>>>(
      Xf16, 0, 512, Wf16, 0, 512, P, 0, 2560, nullptr, 0, 0, 512, 0.f);

  _Float16* a  = P;
  _Float16* bq = P + 512;
  _Float16* cq = P + 1024;
  _Float16* v1 = P + 1536;
  _Float16* v2 = P + 2048;

  // K2: exp'd score matrices
  gemm_wmma<2, false, true><<<dim3(32, 32, 4), blk, 0, stream>>>(
      a,  SP, 2560, bq, SP, 2560, Xm, SN, 2048, nullptr, 0, 0, 512, scale);
  gemm_wmma<2, false, true><<<dim3(32, 32, 4), blk, 0, stream>>>(
      bq, SP, 2560, cq, SP, 2560, Ym, SN, 2048, nullptr, 0, 0, 512, scale);
  gemm_wmma<2, false, true><<<dim3(32, 32, 4), blk, 0, stream>>>(
      cq, SP, 2560, a,  SP, 2560, Zm, SN, 2048, nullptr, 0, 0, 512, scale);

  // K3a: M1T[j,i] = (Y@Z)[j,i] * X[i,j]
  gemm_wmma<3, false, false><<<dim3(32, 32, 4), blk, 0, stream>>>(
      Ym, SN, 2048, Zm, SN, 2048, M1T, SN, 2048, Xm, SN, 2048, 2048, 0.f);
  // K3b: M2[i,k] = (X@Y)[i,k] * Z[k,i]
  gemm_wmma<3, false, false><<<dim3(32, 32, 4), blk, 0, stream>>>(
      Xm, SN, 2048, Ym, SN, 2048, M2, SN, 2048, Zm, SN, 2048, 2048, 0.f);

  // K4d: den = column sums of M1T
  den_kernel<<<dim3(8, 4), dim3(256), 0, stream>>>(M1T, den);

  // K4a: U[:, :512] = (M1T^T @ v1) / den
  gemm_wmma<4, true, false><<<dim3(8, 32, 4), blk, 0, stream>>>(
      M1T, SN, 2048, v1, SP, 2560, U, 2048LL * 1024, 1024, den, 2048, 0, 2048, 0.f);
  // K4b: U[:, 512:] = (M2 @ v2) / den
  gemm_wmma<4, false, false><<<dim3(8, 32, 4), blk, 0, stream>>>(
      M2, SN, 2048, v2, SP, 2560, U + 512, 2048LL * 1024, 1024, den, 2048, 0, 2048, 0.f);

  // K5: out = U @ Ow^T + bias
  gemm_wmma<0, false, true><<<dim3(8, 128, 1), blk, 0, stream>>>(
      U, 0, 1024, OWf16, 0, 1024, out, 0, 512, ob, 0, 0, 1024, 0.f);
}
